// TLSTM3Cell_1331439862441
// MI455X (gfx1250) — matrix operand order repaired and checked by reading, not compiled, not run on hardware
//
#include <hip/hip_runtime.h>
#include <hip/hip_bf16.h>

#define DEV __device__ __forceinline__

// ---------------------------------------------------------------- problem dims
static constexpr int Bsz = 8192;   // batch
static constexpr int Dsz = 512;    // input dim
static constexpr int Hsz = 512;    // hidden dim

// ---------------------------------------------------------------- tiling
static constexpr int BM = 128;     // rows per workgroup (8 waves x 16)
static constexpr int BN = 32;      // cols per workgroup (2 n-tiles of 16)
static constexpr int KT = 32;      // k per WMMA step (bf16 16x16x32)
static constexpr int PITCH = 40;   // LDS row pitch in halves (80B: conflict-free b128)
static constexpr int NTHREADS = 256;

static constexpr int ATILE = BM * PITCH;  // 5120 halves per A plane tile
static constexpr int BTILE = BN * PITCH;  // 1280 halves per B plane tile

// pass1 LDS layout (halves, per buffer): x_hi, x_lo, h_hi, h_lo, 8 weights x {hi,lo}
static constexpr int L1_XHI = 0;
static constexpr int L1_XLO = ATILE;
static constexpr int L1_HHI = 2 * ATILE;
static constexpr int L1_HLO = 3 * ATILE;
static constexpr int L1_B   = 4 * ATILE;
static constexpr int L1_HALVES = 4 * ATILE + 16 * BTILE;   // 40960 per buffer
static constexpr int L1_BYTES  = 2 * L1_HALVES * 2;        // 163840 (double buffered)

// pass2 LDS layout (halves, per buffer): cmt_hi, cmt_lo, Wco {hi,lo}
static constexpr int L2_AHI = 0;
static constexpr int L2_ALO = ATILE;
static constexpr int L2_B   = 2 * ATILE;
static constexpr int L2_HALVES = 2 * ATILE + 2 * BTILE;    // 12800 per buffer
static constexpr int L2_BYTES  = 2 * L2_HALVES * 2;        // 51200

// ---------------------------------------------------------------- types
typedef __attribute__((ext_vector_type(16))) __bf16 v16bf;
typedef __attribute__((ext_vector_type(8)))  float  v8f;

struct alignas(16) B16 { unsigned int x, y, z, w; };
struct alignas(8)  B8  { unsigned int x, y; };
union Frag { v16bf v; B16 q[2]; };

// ---------------------------------------------------------------- helpers
DEV unsigned short bf16_rn(float f) {
  unsigned int u = __float_as_uint(f);
  u += 0x7FFFu + ((u >> 16) & 1u);           // round-to-nearest-even
  return (unsigned short)(u >> 16);
}
DEV float bf16f(unsigned short h) { return __uint_as_float(((unsigned int)h) << 16); }
DEV void split1(float f, unsigned short& hi, unsigned short& lo) {
  hi = bf16_rn(f);
  lo = bf16_rn(f - bf16f(hi));               // residual (bf16x3 split)
}
DEV void split4(const B16& f, B8& hi, B8& lo) {
  unsigned short h0,l0,h1,l1,h2,l2,h3,l3;
  split1(__uint_as_float(f.x), h0, l0);
  split1(__uint_as_float(f.y), h1, l1);
  split1(__uint_as_float(f.z), h2, l2);
  split1(__uint_as_float(f.w), h3, l3);
  hi.x = (unsigned)h0 | ((unsigned)h1 << 16);
  hi.y = (unsigned)h2 | ((unsigned)h3 << 16);
  lo.x = (unsigned)l0 | ((unsigned)l1 << 16);
  lo.y = (unsigned)l2 | ((unsigned)l3 << 16);
}
DEV float sigm(float x) { return 1.0f / (1.0f + expf(-x)); }

DEV v8f v8zero() {
  v8f v;
#pragma unroll
  for (int i = 0; i < 8; ++i) v[i] = 0.0f;
  return v;
}

DEV v8f wmma_b(v16bf a, v16bf b, v8f c) {
  // D = A*B + C, fp32 accumulate; emits v_wmma_f32_16x16x32_bf16
  return __builtin_amdgcn_wmma_f32_16x16x32_bf16(
      /*neg_a*/false, a, /*neg_b*/false, b,
      /*c_mod*/(short)0, c, /*reuse_a*/false, /*reuse_b*/false);
}

// gfx1250 async copy: global memory -> LDS, 16B per enabled lane, ASYNCcnt-tracked.
// ldsAddr is the per-lane LDS *byte* address (low 32 bits of the flat shared addr).
DEV void async_b128(unsigned int ldsAddr, const void* gsrc) {
  asm volatile("global_load_async_to_lds_b128 %0, %1, off"
               :: "v"(ldsAddr), "v"(gsrc)
               : "memory");
}
DEV void wait_async0() {
  asm volatile("s_wait_asynccnt 0" ::: "memory");
}

// A fragment (16x32 bf16, ISA 7.12.2): lane r(<16): K0-7 in v0-3, K16-23 in v4-7;
// lane r+16: K8-15 / K24-31.  Rows are row-major in LDS with pitch PITCH halves.
DEV v16bf loadA(const unsigned short* base, int rowBase, int r, int hi8) {
  const unsigned short* p = base + (size_t)(rowBase + r) * PITCH + hi8 * 8;
  Frag f;
  f.q[0] = *(const B16*)(p);
  f.q[1] = *(const B16*)(p + 16);
  return f.v;
}
// B fragment (32x16 bf16), B = W^T with W row-major [N][K]:
// lane n(<16): K0-7 in v0-3, K16-23 in v4-7; lane n+16: K8-15 / K24-31.
DEV v16bf loadB(const unsigned short* base, int ntBase, int n, int hi8) {
  const unsigned short* p = base + (size_t)(ntBase + n) * PITCH + hi8 * 8;
  Frag f;
  f.q[0] = *(const B16*)(p);
  f.q[1] = *(const B16*)(p + 16);
  return f.v;
}

// ---------------------------------------------------------------- pass 0: split weights
struct SplitArgs { const float* src[9]; };

__global__ __launch_bounds__(256)
void tlstm_split_w(SplitArgs a, unsigned short* hi, unsigned short* lo) {
  const int w = blockIdx.y;                               // 9 weight matrices
  const int i = blockIdx.x * 256 + threadIdx.x;           // 512*512 elements
  const size_t idx = (size_t)w * (Dsz * Hsz) + i;
  unsigned short h, l;
  split1(a.src[w][i], h, l);
  hi[idx] = h;
  lo[idx] = l;
}

// ---------------------------------------------------------------- pass 1
struct P1 {
  const float *x, *dt, *hid, *cs;
  const float *bii, *bhi, *big, *bhg, *bio, *bho, *bto, *bit1, *bit2;
  const float *Wto, *Wtt1, *Wtt2;
  const unsigned short *whi, *wlo;   // 9 planar bf16 weights; pass1 uses [0..7]
  float *preo;
  unsigned short *cmtHi, *cmtLo;
  float *outCm;
};

__global__ __launch_bounds__(NTHREADS)
void tlstm_pass1(P1 A) {
  extern __shared__ unsigned short smem[];
  const unsigned smemBase = (unsigned)(uintptr_t)smem;   // LDS byte address
  const int t       = threadIdx.x;
  const int lane    = t & 31;
  const int wv      = t >> 5;          // 8 waves, each owns 16 rows
  const int r       = lane & 15;
  const int hi8     = lane >> 4;
  const int rowBase = blockIdx.y * BM;
  const int colBase = blockIdx.x * BN;

  // stage A tiles (x, hidden) with on-the-fly bf16x3 split -> LDS buffer `buf`
  auto stageA = [&](int kt, int buf) {
    unsigned short* dst = smem + buf * L1_HALVES;
#pragma unroll
    for (int i = 0; i < 4; ++i) {
      int c = t + i * NTHREADS;        // 1024 float4 chunks per source
      int row = c >> 3, seg = c & 7;
      const float* xp = A.x   + (size_t)(rowBase + row) * Dsz + kt + seg * 4;
      const float* hp = A.hid + (size_t)(rowBase + row) * Hsz + kt + seg * 4;
      if (kt + KT < Dsz) {
        __builtin_prefetch((const void*)(xp + KT), 0, 0);  // global_prefetch_b8
        __builtin_prefetch((const void*)(hp + KT), 0, 0);
      }
      B16 fx = *(const B16*)xp;
      B16 fh = *(const B16*)hp;
      B8 xh, xl, hh, hl;
      split4(fx, xh, xl);
      split4(fh, hh, hl);
      int o = row * PITCH + seg * 4;
      *(B8*)(dst + L1_XHI + o) = xh;
      *(B8*)(dst + L1_XLO + o) = xl;
      *(B8*)(dst + L1_HHI + o) = hh;
      *(B8*)(dst + L1_HLO + o) = hl;
    }
  };
  // stage B tiles: async copy of pre-split bf16 weight planes (8 weights x hi/lo)
  auto stageB = [&](int kt, int buf) {
    const unsigned dstBase = smemBase + (unsigned)(buf * L1_HALVES + L1_B) * 2u;
#pragma unroll
    for (int i = 0; i < 8; ++i) {
      int c = t + i * NTHREADS;        // 2048 chunks of 16B
      int pt  = c >> 7;                // plane-tile 0..15 (weight*2 + plane)
      int row = (c >> 2) & 31;
      int seg = c & 3;
      int w = pt >> 1, lp = pt & 1;
      const unsigned short* src = (lp ? A.wlo : A.whi)
          + (size_t)w * (Dsz * Hsz) + (size_t)(colBase + row) * Dsz + kt + seg * 8;
      async_b128(dstBase + (unsigned)(pt * BTILE + row * PITCH + seg * 8) * 2u, src);
    }
  };

  // acc[o][nt]: o = {t1, t2, i, g, o}
  v8f acc[5][2];
#pragma unroll
  for (int o = 0; o < 5; ++o)
#pragma unroll
    for (int nt = 0; nt < 2; ++nt) acc[o][nt] = v8zero();

  stageA(0, 0);
  stageB(0, 0);
  wait_async0();
  __syncthreads();

  static constexpr int NSTEP = Dsz / KT;   // 16
#pragma unroll 1
  for (int step = 0; step < NSTEP; ++step) {
    const int kt  = step * KT;
    const int cur = step & 1;
    // issue next tile's staging first: async copies overlap the WMMAs below
    if (kt + KT < Dsz) {
      stageB(kt + KT, cur ^ 1);
      stageA(kt + KT, cur ^ 1);
    }

    const unsigned short* base = smem + cur * L1_HALVES;
    v16bf axh = loadA(base + L1_XHI, wv * 16, r, hi8);
    v16bf axl = loadA(base + L1_XLO, wv * 16, r, hi8);
    v16bf ahh = loadA(base + L1_HHI, wv * 16, r, hi8);
    v16bf ahl = loadA(base + L1_HLO, wv * 16, r, hi8);

#pragma unroll
    for (int nt = 0; nt < 2; ++nt) {
#pragma unroll
      for (int w = 0; w < 5; ++w) {    // x-weights: Wit1, Wit2, Wii, Wig, Wio
        v16bf bh = loadB(base + L1_B + (2 * w + 0) * BTILE, nt * 16, r, hi8);
        v16bf bl = loadB(base + L1_B + (2 * w + 1) * BTILE, nt * 16, r, hi8);
        acc[w][nt] = wmma_b(axh, bh, acc[w][nt]);
        acc[w][nt] = wmma_b(axl, bh, acc[w][nt]);
        acc[w][nt] = wmma_b(axh, bl, acc[w][nt]);
      }
#pragma unroll
      for (int w = 0; w < 3; ++w) {    // h-weights: Whi->i, Whg->g, Who->o
        v16bf bh = loadB(base + L1_B + (2 * (5 + w) + 0) * BTILE, nt * 16, r, hi8);
        v16bf bl = loadB(base + L1_B + (2 * (5 + w) + 1) * BTILE, nt * 16, r, hi8);
        const int o = 2 + w;
        acc[o][nt] = wmma_b(ahh, bh, acc[o][nt]);
        acc[o][nt] = wmma_b(ahl, bh, acc[o][nt]);
        acc[o][nt] = wmma_b(ahh, bl, acc[o][nt]);
      }
    }

    wait_async0();       // next-tile async copies landed in LDS
    __syncthreads();     // all waves done reading `cur`, copies visible to all
  }

  // ---- epilogue: nonlinearities + cm / cm_t / pre_o
#pragma unroll
  for (int nt = 0; nt < 2; ++nt) {
    const int col = colBase + nt * 16 + r;
    const float bit1c = A.bit1[col], wt1c = A.Wtt1[col];
    const float bit2c = A.bit2[col], wt2c = A.Wtt2[col];
    const float biic = A.bii[col] + A.bhi[col];
    const float bgc  = A.big[col] + A.bhg[col];
    const float boc  = A.bio[col] + A.bho[col] + A.bto[col];
    const float wtoc = A.Wto[col];
#pragma unroll
    for (int j = 0; j < 8; ++j) {      // C/D layout: M = j + 8*hi8 (ISA 7.12.2)
      const int row = rowBase + wv * 16 + hi8 * 8 + j;
      const size_t idx = (size_t)row * Hsz + col;
      const float dtv = A.dt[row];
      const float cs  = A.cs[idx];
      const float tm1 = sigm(acc[0][nt][j] + bit1c + sigm(dtv * wt1c));
      const float tm2 = sigm(acc[1][nt][j] + bit2c + sigm(dtv * wt2c));
      const float im  = sigm(acc[2][nt][j] + biic);
      const float cur = tanhf(acc[3][nt][j] + bgc);
      const float it1 = im * tm1;
      const float cmt = (1.0f - it1) * cs + it1 * cur;
      const float cm  = (1.0f - im) * cs + im * tm2 * cur;
      A.outCm[idx] = cm;
      A.preo[idx]  = acc[4][nt][j] + boc + dtv * wtoc;
      unsigned short h, l;
      split1(cmt, h, l);               // cm_t as bf16 hi/lo planes for pass 2
      A.cmtHi[idx] = h;
      A.cmtLo[idx] = l;
    }
  }
}

// ---------------------------------------------------------------- pass 2
struct P2 {
  const unsigned short *cmtHi, *cmtLo;  // [B][H] planes
  const unsigned short *wcoHi, *wcoLo;  // [H][H]
  const float *preo, *bco;
  float *outHm;
};

__global__ __launch_bounds__(NTHREADS)
void tlstm_pass2(P2 A) {
  extern __shared__ unsigned short smem[];
  const unsigned smemBase = (unsigned)(uintptr_t)smem;
  const int t       = threadIdx.x;
  const int lane    = t & 31;
  const int wv      = t >> 5;
  const int r       = lane & 15;
  const int hi8     = lane >> 4;
  const int rowBase = blockIdx.y * BM;
  const int colBase = blockIdx.x * BN;

  // everything here is a pure bf16 copy -> fully async staging
  auto stage = [&](int kt, int buf) {
    const unsigned dstBase = smemBase + (unsigned)(buf * L2_HALVES) * 2u;
#pragma unroll
    for (int i = 0; i < 4; ++i) {
      int c = t + i * NTHREADS;        // 1024 chunks: cm_t hi/lo planes
      int pl = c >> 9, row = (c >> 2) & 127, seg = c & 3;
      const unsigned short* src = (pl ? A.cmtLo : A.cmtHi)
          + (size_t)(rowBase + row) * Hsz + kt + seg * 8;
      async_b128(dstBase + (unsigned)((pl ? L2_ALO : L2_AHI) + row * PITCH + seg * 8) * 2u,
                 src);
    }
    if (t < 128) {                     // Wco hi/lo tiles
      int pl = t >> 6, row = (t >> 2) & 31, seg = t & 3;
      const unsigned short* src = (pl ? A.wcoLo : A.wcoHi)
          + (size_t)(colBase + row) * Hsz + kt + seg * 8;
      async_b128(dstBase + (unsigned)(L2_B + pl * BTILE + row * PITCH + seg * 8) * 2u,
                 src);
    }
  };

  v8f acc[2];
  acc[0] = v8zero();
  acc[1] = v8zero();

  stage(0, 0);
  wait_async0();
  __syncthreads();

  static constexpr int NSTEP = Hsz / KT;
#pragma unroll 1
  for (int step = 0; step < NSTEP; ++step) {
    const int kt  = step * KT;
    const int cur = step & 1;
    if (kt + KT < Hsz) stage(kt + KT, cur ^ 1);

    const unsigned short* base = smem + cur * L2_HALVES;
    v16bf ah = loadA(base + L2_AHI, wv * 16, r, hi8);
    v16bf al = loadA(base + L2_ALO, wv * 16, r, hi8);
#pragma unroll
    for (int nt = 0; nt < 2; ++nt) {
      v16bf bh = loadB(base + L2_B + 0 * BTILE, nt * 16, r, hi8);
      v16bf bl = loadB(base + L2_B + 1 * BTILE, nt * 16, r, hi8);
      acc[nt] = wmma_b(ah, bh, acc[nt]);
      acc[nt] = wmma_b(al, bh, acc[nt]);
      acc[nt] = wmma_b(ah, bl, acc[nt]);
    }

    wait_async0();
    __syncthreads();
  }

#pragma unroll
  for (int nt = 0; nt < 2; ++nt) {
    const int col = colBase + nt * 16 + r;
    const float bcoc = A.bco[col];
#pragma unroll
    for (int j = 0; j < 8; ++j) {
      const int row = rowBase + wv * 16 + hi8 * 8 + j;
      const size_t idx = (size_t)row * Hsz + col;
      const float om  = sigm(A.preo[idx] + acc[nt][j] + bcoc);
      const float cmt = bf16f(A.cmtHi[idx]) + bf16f(A.cmtLo[idx]);
      A.outHm[idx] = om * tanhf(cmt);
    }
  }
}

// ---------------------------------------------------------------- host
extern "C" void kernel_launch(void* const* d_in, const int* in_sizes, int n_in,
                              void* d_out, int out_size, void* d_ws, size_t ws_size,
                              hipStream_t stream) {
  (void)in_sizes; (void)n_in; (void)out_size; (void)ws_size;

  const float* x   = (const float*)d_in[0];
  const float* dt  = (const float*)d_in[1];
  const float* hid = (const float*)d_in[2];
  const float* cs  = (const float*)d_in[3];
  const float* Wii = (const float*)d_in[4];   const float* bii = (const float*)d_in[5];
  const float* Whi = (const float*)d_in[6];   const float* bhi = (const float*)d_in[7];
  const float* Wig = (const float*)d_in[8];   const float* big = (const float*)d_in[9];
  const float* Whg = (const float*)d_in[10];  const float* bhg = (const float*)d_in[11];
  const float* Wio = (const float*)d_in[12];  const float* bio = (const float*)d_in[13];
  const float* Who = (const float*)d_in[14];  const float* bho = (const float*)d_in[15];
  const float* Wto = (const float*)d_in[16];  const float* bto = (const float*)d_in[17];
  const float* Wco = (const float*)d_in[18];  const float* bco = (const float*)d_in[19];
  const float* Wit1 = (const float*)d_in[20]; const float* bit1 = (const float*)d_in[21];
  const float* Wtt1 = (const float*)d_in[22];
  const float* Wit2 = (const float*)d_in[23]; const float* bit2 = (const float*)d_in[24];
  const float* Wtt2 = (const float*)d_in[25];

  float* outHm = (float*)d_out;
  float* outCm = outHm + (size_t)Bsz * Hsz;

  // workspace layout (~41 MB total)
  const size_t WELEMS = (size_t)Dsz * Hsz;            // 262144 per weight
  char* ws = (char*)d_ws;
  unsigned short* whi   = (unsigned short*)ws;                    // 9 * 512KB
  unsigned short* wlo   = whi + 9 * WELEMS;                       // 9 * 512KB
  float*          preo  = (float*)(ws + 2 * 9 * WELEMS * 2);      // 16 MB
  unsigned short* cmtHi = (unsigned short*)((char*)preo + (size_t)Bsz * Hsz * 4);
  unsigned short* cmtLo = cmtHi + (size_t)Bsz * Hsz;

  (void)hipFuncSetAttribute(reinterpret_cast<const void*>(tlstm_pass1),
                            hipFuncAttributeMaxDynamicSharedMemorySize, L1_BYTES);
  (void)hipFuncSetAttribute(reinterpret_cast<const void*>(tlstm_pass2),
                            hipFuncAttributeMaxDynamicSharedMemorySize, L2_BYTES);

  // pass 0: split the 9 weight matrices to bf16 hi/lo planes
  SplitArgs sa;
  sa.src[0] = Wit1; sa.src[1] = Wit2; sa.src[2] = Wii; sa.src[3] = Wig; sa.src[4] = Wio;
  sa.src[5] = Whi;  sa.src[6] = Whg;  sa.src[7] = Who; sa.src[8] = Wco;
  tlstm_split_w<<<dim3(WELEMS / 256, 9), 256, 0, stream>>>(sa, whi, wlo);

  // pass 1: fused 8-GEMM (bf16x3) + gate nonlinearities
  P1 p1;
  p1.x = x; p1.dt = dt; p1.hid = hid; p1.cs = cs;
  p1.bii = bii; p1.bhi = bhi; p1.big = big; p1.bhg = bhg;
  p1.bio = bio; p1.bho = bho; p1.bto = bto; p1.bit1 = bit1; p1.bit2 = bit2;
  p1.Wto = Wto; p1.Wtt1 = Wtt1; p1.Wtt2 = Wtt2;
  p1.whi = whi; p1.wlo = wlo;
  p1.preo = preo; p1.cmtHi = cmtHi; p1.cmtLo = cmtLo; p1.outCm = outCm;
  tlstm_pass1<<<dim3(Hsz / BN, Bsz / BM), NTHREADS, L1_BYTES, stream>>>(p1);

  // pass 2: cm_t @ Wco^T (bf16x3) + om/hm epilogue
  P2 p2;
  p2.cmtHi = cmtHi; p2.cmtLo = cmtLo;
  p2.wcoHi = whi + 8 * WELEMS; p2.wcoLo = wlo + 8 * WELEMS;
  p2.preo = preo; p2.bco = bco; p2.outHm = outHm;
  tlstm_pass2<<<dim3(Hsz / BN, Bsz / BM), NTHREADS, L2_BYTES, stream>>>(p2);
}
